// GraphAttentionKAN_58626303591105
// MI455X (gfx1250) — matrix operand order repaired and checked by reading, not compiled, not run on hardware
//
#include <hip/hip_runtime.h>
#include <math.h>

// ---------------- problem constants (from reference setup_inputs) ----------
static constexpr int cN   = 20000;
static constexpr int cE   = 160000;
static constexpr int cF   = 64;
static constexpr int cIDE = 32;
static constexpr int cCAT = cF + cIDE;   // 96
static constexpr int cHID = 128;
static constexpr int cH   = 4;
static constexpr int cHC  = 512;         // H * HID
static constexpr int cC   = cHC / cH;    // 128 per head
static constexpr int cR   = 4;
static constexpr int cL   = 3;
static constexpr int cED  = 24;          // 16 edge_attr + 8 rel_emb
static constexpr int cFFN = 256;
static constexpr int cSK  = 8;           // GRID+ORDER
static constexpr int cKH  = 128;
static constexpr int cNC  = 10;
static constexpr int cNG  = 64;

typedef __attribute__((ext_vector_type(16))) __bf16 v16bf;
typedef __attribute__((ext_vector_type(8)))  float  v8f;

__device__ __forceinline__ float siluf(float x) { return x / (1.0f + __expf(-x)); }

__device__ __forceinline__ void atomicMaxF(float* addr, float val) {
  unsigned int* ua = (unsigned int*)addr;
  unsigned int old = __float_as_uint(*addr);
  while (__uint_as_float(old) < val) {
    unsigned int prev = atomicCAS(ua, old, __float_as_uint(val));
    if (prev == old) break;
    old = prev;
  }
}

// ===========================================================================
// Tiled WMMA GEMM:  C[M,Nn] = epi( act_a(A[M,K]) * B[Nn,K]^T + bias )
//   ACT_A: 0 = none, 1 = silu(A)
//   EPI  : 0 = store (+bias), 1 = silu(out+bias), 2 = C += out (accumulate)
//   HB   : bias present
// REQUIRES: K % 32 == 0 (true for every call site here).
// Block = 256 threads (8 waves), tile 128x128, BK=32; each wave owns a 32x64
// quadrant = 2x4 v_wmma_f32_16x16x32_bf16 accumulators.
// Software pipeline: commit(regs->LDS, buf cur) | barrier | issue next global
// loads (128-bit, coalesced, row-clamped for tails) | ds_load frags + 8 wmma.
// Tail rows are clamped but NOT zeroed: polluted accumulator rows/cols are
// exactly those the guarded epilogue never stores, K is always fully valid.
// ===========================================================================
template<int ACT_A, int EPI, int HB>
__global__ __launch_bounds__(256)
void gemm_bf16_wmma(const float* __restrict__ A, const float* __restrict__ B,
                    const float* __restrict__ bias, float* __restrict__ C,
                    int M, int Nn, int K)
{
  constexpr int BM = 128, BN = 128, BK = 32, LDT = BK + 8; // LDS row stride (bf16)
  __shared__ __bf16 As[2][BM * LDT];
  __shared__ __bf16 Bs[2][BN * LDT];

  const int tid  = threadIdx.x;
  const int wave = tid >> 5;
  const int lane = tid & 31;
  const int half = lane >> 4;
  const int l16  = lane & 15;
  const int wm   = (wave >> 1) * 32;   // 0,32,64,96
  const int wn   = (wave & 1) * 64;    // 0,64
  const int bm   = blockIdx.y * BM;
  const int bn   = blockIdx.x * BN;

  // per-thread staging coordinates: 128x32 floats = 1024 float4 per matrix,
  // 256 threads -> 4 float4 each.
  int srow[4], scol[4];
  const float* aptr[4];
  const float* bptr[4];
#pragma unroll
  for (int it = 0; it < 4; ++it) {
    int idx = tid + it * 256;
    srow[it] = idx >> 3;
    scol[it] = (idx & 7) * 4;
    int ga = bm + srow[it]; ga = ga < M ? ga : (M - 1);    // clamp, no zeroing
    int gb = bn + srow[it]; gb = gb < Nn ? gb : (Nn - 1);
    aptr[it] = A + (size_t)ga * K + scol[it];
    bptr[it] = B + (size_t)gb * K + scol[it];
  }

  v8f acc[2][4] = {};
  float4 ra[4], rb[4];

  auto issue_loads = [&](int k0) {
#pragma unroll
    for (int it = 0; it < 4; ++it) ra[it] = *(const float4*)(aptr[it] + k0);
#pragma unroll
    for (int it = 0; it < 4; ++it) rb[it] = *(const float4*)(bptr[it] + k0);
  };
  auto commit = [&](int buf) {
#pragma unroll
    for (int it = 0; it < 4; ++it) {
      float x0 = ra[it].x, x1 = ra[it].y, x2 = ra[it].z, x3 = ra[it].w;
      if (ACT_A == 1) { x0 = siluf(x0); x1 = siluf(x1); x2 = siluf(x2); x3 = siluf(x3); }
      __bf16* d = &As[buf][srow[it] * LDT + scol[it]];
      d[0] = (__bf16)x0; d[1] = (__bf16)x1; d[2] = (__bf16)x2; d[3] = (__bf16)x3;
    }
#pragma unroll
    for (int it = 0; it < 4; ++it) {
      __bf16* d = &Bs[buf][srow[it] * LDT + scol[it]];
      d[0] = (__bf16)rb[it].x; d[1] = (__bf16)rb[it].y;
      d[2] = (__bf16)rb[it].z; d[3] = (__bf16)rb[it].w;
    }
  };

  issue_loads(0);
  int cur = 0;
  for (int k0 = 0; k0 < K; k0 += BK) {
    commit(cur);                 // waits on in-flight loads, packs to LDS
    __syncthreads();             // buf[cur] visible; buf[cur^1] free
    if (k0 + BK < K) issue_loads(k0 + BK);   // next tile in flight during wmma

    v16bf af[2], bf[4];
#pragma unroll
    for (int s = 0; s < 2; ++s) {
      const __bf16* ap = &As[cur][(wm + s * 16 + l16) * LDT];
#pragma unroll
      for (int j = 0; j < 8; ++j) {
        // A 16x32 bf16 layout (ISA 7.12.2): half-wave K {0..7,16..23} / {8..15,24..31}
        int ka = (j < 4 ? 2 * j : 16 + 2 * (j - 4)) + 8 * half;
        af[s][2 * j]     = ap[ka];
        af[s][2 * j + 1] = ap[ka + 1];
      }
    }
#pragma unroll
    for (int t = 0; t < 4; ++t) {
      const __bf16* bp = &Bs[cur][(wn + t * 16 + l16) * LDT];
#pragma unroll
      for (int j = 0; j < 8; ++j) {
        // B 32x16 bf16 layout: half-wave K {0..7,16..23} / {8..15,24..31} (matches A)
        int kb = (j < 4 ? 2 * j : 16 + 2 * (j - 4)) + 8 * half;
        bf[t][2 * j]     = bp[kb];
        bf[t][2 * j + 1] = bp[kb + 1];
      }
    }
#pragma unroll
    for (int i = 0; i < 2; ++i)
#pragma unroll
      for (int j = 0; j < 4; ++j)
        acc[i][j] = __builtin_amdgcn_wmma_f32_16x16x32_bf16(
            false, af[i], false, bf[j], (short)0, acc[i][j], false, false);
    cur ^= 1;
  }

  // C/D layout: row = vgpr + 8*half, col = lane&15
#pragma unroll
  for (int i = 0; i < 2; ++i) {
#pragma unroll
    for (int j = 0; j < 4; ++j) {
      int col = bn + wn + j * 16 + l16;
      if (col >= Nn) continue;
      float bv = HB ? bias[col] : 0.0f;
#pragma unroll
      for (int v = 0; v < 8; ++v) {
        int row = bm + wm + i * 16 + v + 8 * half;
        if (row < M) {
          float o = acc[i][j][v] + bv;
          if (EPI == 1) o = siluf(o);
          size_t idx = (size_t)row * Nn + col;
          if (EPI == 2) C[idx] += o; else C[idx] = o;
        }
      }
    }
  }
}

// ---------------- misc elementwise kernels ---------------------------------
__global__ void fill_f32_kernel(float* p, float v, long n) {
  long i = (long)blockIdx.x * 256 + threadIdx.x;
  if (i < n) p[i] = v;
}

__global__ void build_hcat_kernel(const float* __restrict__ x,
                                  const float* __restrict__ id_emb,
                                  const int* __restrict__ id_token,
                                  float* __restrict__ hcat) {
  long i = (long)blockIdx.x * 256 + threadIdx.x;
  if (i >= (long)cN * cCAT) return;
  int n = (int)(i / cCAT), c = (int)(i % cCAT);
  hcat[i] = (c < cF) ? x[(size_t)n * cF + c]
                     : id_emb[(size_t)id_token[n] * cIDE + (c - cF)];
}

// sw[o,d,s] = spline_w[o,d,s] * scaler[o,d]   (same flat layout as reshape)
__global__ void make_sw_kernel(const float* __restrict__ sp, const float* __restrict__ sc,
                               float* __restrict__ sw, int outd, int din) {
  long i = (long)blockIdx.x * 256 + threadIdx.x;
  long tot = (long)outd * din * cSK;
  if (i >= tot) return;
  long rest = i >> 3;
  int d = (int)(rest % din), o = (int)(rest / din);
  sw[i] = sp[i] * sc[(size_t)o * din + d];
}

// B-spline bases (uniform grid G=5,k=3, lo=-1, h=0.4): per (row,feature) -> 8 vals
__global__ void spline_basis_kernel(const float* __restrict__ x, float* __restrict__ basis,
                                    long total) {
  long i = (long)blockIdx.x * 256 + threadIdx.x;
  if (i >= total) return;
  const float lo = -1.0f, hs = 0.4f;
  float xv = x[i];
  float b[11];
#pragma unroll
  for (int j = 0; j < 11; ++j) {
    float gj = lo + (j - 3) * hs;
    b[j] = (xv >= gj && xv < gj + hs) ? 1.0f : 0.0f;
  }
#pragma unroll
  for (int k = 1; k <= 3; ++k) {
    for (int j = 0; j + k < 11; ++j) {
      float gj  = lo + (j - 3) * hs;
      float gj1 = gj + hs;
      float gjk = gj + k * hs;
      float gk1 = gj1 + k * hs;
      b[j] = (xv - gj) / (gjk - gj) * b[j] + (gk1 - xv) / (gk1 - gj1) * b[j + 1];
    }
  }
  float* o = basis + i * cSK;
#pragma unroll
  for (int j = 0; j < cSK; ++j) o[j] = b[j];
}

// ---------------- graph attention kernels (wave per edge) ------------------
__global__ __launch_bounds__(256)
void edge_alpha_kernel(const float* __restrict__ xl, const float* __restrict__ xr,
                       const float* __restrict__ edge_attr,
                       const float* __restrict__ rel_emb_l,   // [R,8]
                       const float* __restrict__ lin_e_w_l,   // [R,HC,ED]
                       const float* __restrict__ att_l,       // [R,H,128]
                       const int* __restrict__ src, const int* __restrict__ dst,
                       const int* __restrict__ etype, float* __restrict__ alpha) {
  int e = blockIdx.x * 8 + (threadIdx.x >> 5);
  if (e >= cE) return;
  int lane = threadIdx.x & 31;
  int r = etype[e], s = src[e], d = dst[e];
  float ea[cED];
#pragma unroll
  for (int i = 0; i < 16; ++i) ea[i] = edge_attr[(size_t)e * 16 + i];
#pragma unroll
  for (int i = 0; i < 8; ++i) ea[16 + i] = rel_emb_l[r * 8 + i];
  const float* xlp = xl + ((size_t)r * cN + s) * cHC + lane * 16;
  const float* xrp = xr + ((size_t)r * cN + d) * cHC + lane * 16;
  const float* ew  = lin_e_w_l + ((size_t)r * cHC + lane * 16) * cED;
  int head = lane >> 3;
  const float* attp = att_l + ((size_t)r * cH + head) * cC;
  float partial = 0.0f;
#pragma unroll
  for (int c = 0; c < 16; ++c) {
    float eemb = 0.0f;
#pragma unroll
    for (int kk = 0; kk < cED; ++kk) eemb += ew[c * cED + kk] * ea[kk];
    float m = xlp[c] + xrp[c] + eemb;
    m = m > 0.0f ? m : 0.2f * m;                 // leaky_relu 0.2
    partial += m * attp[(lane * 16 + c) & (cC - 1)];
  }
  partial += __shfl_xor(partial, 1, 32);
  partial += __shfl_xor(partial, 2, 32);
  partial += __shfl_xor(partial, 4, 32);
  if ((lane & 7) == 0) alpha[(size_t)e * cH + head] = partial;
}

__global__ void seg_max_kernel(const float* __restrict__ alpha, const int* __restrict__ dst,
                               const int* __restrict__ etype, float* __restrict__ amax) {
  long i = (long)blockIdx.x * 256 + threadIdx.x;
  if (i >= (long)cE * cH) return;
  int e = (int)(i >> 2), h = (int)(i & 3);
  int seg = dst[e] * cR + etype[e];
  atomicMaxF(&amax[(size_t)seg * cH + h], alpha[i]);
}

__global__ void seg_exp_sum_kernel(const float* __restrict__ alpha, const int* __restrict__ dst,
                                   const int* __restrict__ etype,
                                   const float* __restrict__ amax,
                                   float* __restrict__ ex, float* __restrict__ den) {
  long i = (long)blockIdx.x * 256 + threadIdx.x;
  if (i >= (long)cE * cH) return;
  int e = (int)(i >> 2), h = (int)(i & 3);
  int seg = dst[e] * cR + etype[e];
  float v = __expf(alpha[i] - amax[(size_t)seg * cH + h]);
  ex[i] = v;
  atomicAdd(&den[(size_t)seg * cH + h], v);
}

__global__ __launch_bounds__(256)
void agg_kernel(const float* __restrict__ xl, const float* __restrict__ ex,
                const float* __restrict__ den,
                const int* __restrict__ src, const int* __restrict__ dst,
                const int* __restrict__ etype, float* __restrict__ agg) {
  int e = blockIdx.x * 8 + (threadIdx.x >> 5);
  if (e >= cE) return;
  int lane = threadIdx.x & 31;
  int r = etype[e], s = src[e];
  int seg = dst[e] * cR + r;
  int head = lane >> 3;
  float a = ex[(size_t)e * cH + head] / den[(size_t)seg * cH + head];
  const float* xlp = xl + ((size_t)r * cN + s) * cHC + lane * 16;
  float* ag = agg + (size_t)seg * cHC + lane * 16;
#pragma unroll
  for (int c = 0; c < 16; ++c) atomicAdd(&ag[c], a * xlp[c]);
}

// h = LN( h + sum_r softmax(rel_gate)[r] * (mean_heads(agg[n,r]) + gat_bias[r]) )
__global__ __launch_bounds__(256)
void node_update_kernel(float* __restrict__ h, const float* __restrict__ agg,
                        const float* __restrict__ gat_bias_l,  // [R,128]
                        const float* __restrict__ rel_gate_l,  // [R]
                        const float* __restrict__ w, const float* __restrict__ b) {
  int n = blockIdx.x * 8 + (threadIdx.x >> 5);
  if (n >= cN) return;
  int lane = threadIdx.x & 31;
  float g[cR];
  float gm = -1e30f;
#pragma unroll
  for (int r = 0; r < cR; ++r) { g[r] = rel_gate_l[r]; gm = fmaxf(gm, g[r]); }
  float gs = 0.0f;
#pragma unroll
  for (int r = 0; r < cR; ++r) { g[r] = __expf(g[r] - gm); gs += g[r]; }
#pragma unroll
  for (int r = 0; r < cR; ++r) g[r] /= gs;

  float t[4], s = 0.0f, s2 = 0.0f;
#pragma unroll
  for (int i = 0; i < 4; ++i) {
    int c = lane + 32 * i;
    float hm = 0.0f;
#pragma unroll
    for (int r = 0; r < cR; ++r) {
      const float* ap = agg + ((size_t)(n * cR + r)) * cHC + c;
      float o = 0.25f * (ap[0] + ap[cC] + ap[2 * cC] + ap[3 * cC]) + gat_bias_l[r * cHID + c];
      hm += g[r] * o;
    }
    float v = h[(size_t)n * cHID + c] + hm;
    t[i] = v; s += v; s2 += v * v;
  }
  for (int off = 16; off; off >>= 1) { s += __shfl_xor(s, off, 32); s2 += __shfl_xor(s2, off, 32); }
  float mu = s / cHID, var = s2 / cHID - mu * mu, inv = rsqrtf(var + 1e-5f);
#pragma unroll
  for (int i = 0; i < 4; ++i) {
    int c = lane + 32 * i;
    h[(size_t)n * cHID + c] = (t[i] - mu) * inv * w[c] + b[c];
  }
}

// h = LN(h + y)  over D=128
__global__ __launch_bounds__(256)
void residual_ln_kernel(float* __restrict__ h, const float* __restrict__ y,
                        const float* __restrict__ w, const float* __restrict__ b) {
  int n = blockIdx.x * 8 + (threadIdx.x >> 5);
  if (n >= cN) return;
  int lane = threadIdx.x & 31;
  float t[4], s = 0.0f, s2 = 0.0f;
#pragma unroll
  for (int i = 0; i < 4; ++i) {
    int c = lane + 32 * i;
    float v = h[(size_t)n * cHID + c] + y[(size_t)n * cHID + c];
    t[i] = v; s += v; s2 += v * v;
  }
  for (int off = 16; off; off >>= 1) { s += __shfl_xor(s, off, 32); s2 += __shfl_xor(s2, off, 32); }
  float mu = s / cHID, var = s2 / cHID - mu * mu, inv = rsqrtf(var + 1e-5f);
#pragma unroll
  for (int i = 0; i < 4; ++i) {
    int c = lane + 32 * i;
    h[(size_t)n * cHID + c] = (t[i] - mu) * inv * w[c] + b[c];
  }
}

// ---------------- readout --------------------------------------------------
__global__ __launch_bounds__(256)
void readout_acc_kernel(const float* __restrict__ h, const int* __restrict__ batch,
                        float* __restrict__ sums, float* __restrict__ mx,
                        int* __restrict__ cnt) {
  int n = blockIdx.x * 8 + (threadIdx.x >> 5);
  if (n >= cN) return;
  int lane = threadIdx.x & 31;
  int g = batch[n];
  if (lane == 0) atomicAdd(&cnt[g], 1);
#pragma unroll
  for (int i = 0; i < 4; ++i) {
    int c = lane + 32 * i;
    float v = h[(size_t)n * cHID + c];
    atomicAdd(&sums[(size_t)g * cHID + c], v);
    atomicMaxF(&mx[(size_t)g * cHID + c], v);
  }
}

// gcat[g] = LN(concat(mean, max)) with ro_norm; one 256-thread block per graph
__global__ __launch_bounds__(256)
void readout_norm_kernel(const float* __restrict__ sums, const float* __restrict__ mx,
                         const int* __restrict__ cnt, const float* __restrict__ w,
                         const float* __restrict__ b, float* __restrict__ gcat) {
  int g = blockIdx.x, t = threadIdx.x;
  float cf = (float)cnt[g];
  float v;
  if (t < cHID) v = sums[(size_t)g * cHID + t] / fmaxf(cf, 1.0f);
  else { float m = mx[(size_t)g * cHID + (t - cHID)]; v = (cf > 0.0f) ? m : 0.0f; }
  float s = v, s2 = v * v;
  for (int off = 16; off; off >>= 1) { s += __shfl_xor(s, off, 32); s2 += __shfl_xor(s2, off, 32); }
  __shared__ float ss[8], ss2[8];
  int wv = t >> 5, lane = t & 31;
  if (lane == 0) { ss[wv] = s; ss2[wv] = s2; }
  __syncthreads();
  if (t == 0) {
    float a = 0, c2 = 0;
    for (int i = 0; i < 8; ++i) { a += ss[i]; c2 += ss2[i]; }
    ss[0] = a; ss2[0] = c2;
  }
  __syncthreads();
  float mu = ss[0] / 256.0f, var = ss2[0] / 256.0f - mu * mu;
  gcat[(size_t)g * 256 + t] = (v - mu) * rsqrtf(var + 1e-5f) * w[t] + b[t];
}

// ---------------- host orchestration ---------------------------------------
template<int ACT_A, int EPI, int HB>
static inline void run_gemm(const float* A, const float* B, const float* bias, float* C,
                            int M, int Nn, int K, hipStream_t s) {
  dim3 grid((Nn + 127) / 128, (M + 127) / 128), block(256);
  gemm_bf16_wmma<ACT_A, EPI, HB><<<grid, block, 0, s>>>(A, B, bias, C, M, Nn, K);
}

static inline long cdiv(long a, long b) { return (a + b - 1) / b; }

extern "C" void kernel_launch(void* const* d_in, const int* in_sizes, int n_in,
                              void* d_out, int out_size, void* d_ws, size_t ws_size,
                              hipStream_t stream) {
  const float* x          = (const float*)d_in[0];
  const float* edge_attr  = (const float*)d_in[1];
  const float* id_emb     = (const float*)d_in[2];
  const float* in_proj_w  = (const float*)d_in[3];
  const float* in_proj_b  = (const float*)d_in[4];
  const float* rel_emb    = (const float*)d_in[5];
  const float* lin_l_w    = (const float*)d_in[6];
  const float* lin_l_b    = (const float*)d_in[7];
  const float* lin_r_w    = (const float*)d_in[8];
  const float* lin_r_b    = (const float*)d_in[9];
  const float* lin_e_w    = (const float*)d_in[10];
  const float* att        = (const float*)d_in[11];
  const float* gat_bias   = (const float*)d_in[12];
  const float* rel_gate   = (const float*)d_in[13];
  const float* norm1_w    = (const float*)d_in[14];
  const float* norm1_b    = (const float*)d_in[15];
  const float* norm2_w    = (const float*)d_in[16];
  const float* norm2_b    = (const float*)d_in[17];
  const float* ffn1_base  = (const float*)d_in[18];
  const float* ffn1_spw   = (const float*)d_in[19];
  const float* ffn1_sc    = (const float*)d_in[20];
  const float* ffn2_base  = (const float*)d_in[21];
  const float* ffn2_spw   = (const float*)d_in[22];
  const float* ffn2_sc    = (const float*)d_in[23];
  const float* ro_norm_w  = (const float*)d_in[24];
  const float* ro_norm_b  = (const float*)d_in[25];
  const float* h1_base    = (const float*)d_in[26];
  const float* h1_spw     = (const float*)d_in[27];
  const float* h1_sc      = (const float*)d_in[28];
  const float* h2_base    = (const float*)d_in[29];
  const float* h2_spw     = (const float*)d_in[30];
  const float* h2_sc      = (const float*)d_in[31];
  const int*   id_token   = (const int*)d_in[32];
  const int*   edge_index = (const int*)d_in[33];
  const int*   edge_type  = (const int*)d_in[34];
  const int*   batch      = (const int*)d_in[35];
  float* out = (float*)d_out;
  (void)in_sizes; (void)n_in; (void)out_size; (void)ws_size;

  const int* src = edge_index;
  const int* dst = edge_index + cE;

  // ---- workspace carving (aligned 256B); big buffers overlaid by lifetime --
  char* ws = (char*)d_ws;
  size_t off = 0;
  auto carve = [&](size_t bytes) -> float* {
    float* p = (float*)(ws + off);
    off = (off + bytes + 255) & ~(size_t)255;
    return p;
  };
  float* h     = carve((size_t)cN * cHID * 4);
  float* hcat  = carve((size_t)cN * cCAT * 4);
  float* xl    = carve((size_t)cR * cN * cHC * 4);   // later: basisF [N, FFN*8]
  float* xr    = carve((size_t)cR * cN * cHC * 4);   // later: agg, then basisH
  float* alpha = carve((size_t)cE * cH * 4);
  float* ex    = carve((size_t)cE * cH * 4);
  float* amax  = carve((size_t)cN * cR * cH * 4);
  float* den   = carve((size_t)cN * cR * cH * 4);
  float* y1    = carve((size_t)cN * cFFN * 4);
  float* y2    = carve((size_t)cN * cHID * 4);
  float* sw1   = carve((size_t)cFFN * cHID * cSK * 4);
  float* sw2   = carve((size_t)cHID * cFFN * cSK * 4);
  int*   cnt   = (int*)carve((size_t)cNG * 4);
  float* sums  = carve((size_t)cNG * cHID * 4);
  float* mx    = carve((size_t)cNG * cHID * 4);
  float* gcat  = carve((size_t)cNG * 256 * 4);
  float* bgin  = carve((size_t)cNG * 256 * cSK * 4);
  float* g2    = carve((size_t)cNG * cKH * 4);
  float* bg2   = carve((size_t)cNG * cKH * cSK * 4);
  float* swh1  = carve((size_t)cKH * 256 * cSK * 4);
  float* swh2  = carve((size_t)cNC * cKH * cSK * 4);
  float* agg    = xr;   // overlay: xr dead once edge_alpha done
  float* basisH = xr;   // overlay: agg dead once node_update done
  float* basisF = xl;   // overlay: xl dead once agg done

  // ---- input projection: h = silu(concat(x, id_emb[id]) @ W^T + b) --------
  build_hcat_kernel<<<cdiv((long)cN * cCAT, 256), 256, 0, stream>>>(x, id_emb, id_token, hcat);
  run_gemm<0, 1, 1>(hcat, in_proj_w, in_proj_b, h, cN, cHID, cCAT, stream);

  for (int l = 0; l < cL; ++l) {
    const float* llw = lin_l_w + (size_t)l * cR * cHC * cHID;
    const float* llb = lin_l_b + (size_t)l * cR * cHC;
    const float* lrw = lin_r_w + (size_t)l * cR * cHC * cHID;
    const float* lrb = lin_r_b + (size_t)l * cR * cHC;
    const float* lew = lin_e_w + (size_t)l * cR * cHC * cED;
    const float* rel = rel_emb + (size_t)l * cR * 8;
    const float* atl = att + (size_t)l * cR * cH * cC;
    const float* gbl = gat_bias + (size_t)l * cR * cHID;
    const float* rgl = rel_gate + (size_t)l * cR;

    // xl[r] = h @ lin_l_w[l,r]^T + b ;  xr[r] likewise (WMMA GEMMs)
    for (int r = 0; r < cR; ++r) {
      run_gemm<0, 0, 1>(h, llw + (size_t)r * cHC * cHID, llb + (size_t)r * cHC,
                        xl + (size_t)r * cN * cHC, cN, cHC, cHID, stream);
      run_gemm<0, 0, 1>(h, lrw + (size_t)r * cHC * cHID, lrb + (size_t)r * cHC,
                        xr + (size_t)r * cN * cHC, cN, cHC, cHID, stream);
    }

    // attention logits
    edge_alpha_kernel<<<cdiv(cE, 8), 256, 0, stream>>>(xl, xr, edge_attr, rel, lew, atl,
                                                       src, dst, edge_type, alpha);
    // segment softmax (dst*R+rel segments) via atomics
    fill_f32_kernel<<<cdiv((long)cN * cR * cH, 256), 256, 0, stream>>>(amax, -1e30f, (long)cN * cR * cH);
    seg_max_kernel<<<cdiv((long)cE * cH, 256), 256, 0, stream>>>(alpha, dst, edge_type, amax);
    hipMemsetAsync(den, 0, (size_t)cN * cR * cH * 4, stream);
    seg_exp_sum_kernel<<<cdiv((long)cE * cH, 256), 256, 0, stream>>>(alpha, dst, edge_type, amax, ex, den);
    // weighted aggregation (agg overlays xr)
    hipMemsetAsync(agg, 0, (size_t)cR * cN * cHC * 4, stream);
    agg_kernel<<<cdiv(cE, 8), 256, 0, stream>>>(xl, ex, den, src, dst, edge_type, agg);
    // head-mean + gated relation mix + residual LN
    node_update_kernel<<<cdiv(cN, 8), 256, 0, stream>>>(h, agg, gbl, rgl,
                                                        norm1_w + l * cHID, norm1_b + l * cHID);

    // ---- KAN FFN (base: silu GEMM; spline: basis GEMM, fused via accumulate)
    const float* f1b = ffn1_base + (size_t)l * cFFN * cHID;
    const float* f2b = ffn2_base + (size_t)l * cHID * cFFN;
    make_sw_kernel<<<cdiv((long)cFFN * cHID * cSK, 256), 256, 0, stream>>>(
        ffn1_spw + (size_t)l * cFFN * cHID * cSK, ffn1_sc + (size_t)l * cFFN * cHID, sw1, cFFN, cHID);
    make_sw_kernel<<<cdiv((long)cHID * cFFN * cSK, 256), 256, 0, stream>>>(
        ffn2_spw + (size_t)l * cHID * cFFN * cSK, ffn2_sc + (size_t)l * cHID * cFFN, sw2, cHID, cFFN);

    spline_basis_kernel<<<cdiv((long)cN * cHID, 256), 256, 0, stream>>>(h, basisH, (long)cN * cHID);
    run_gemm<1, 0, 0>(h, f1b, nullptr, y1, cN, cFFN, cHID, stream);               // silu(h)@base1^T
    run_gemm<0, 2, 0>(basisH, sw1, nullptr, y1, cN, cFFN, cHID * cSK, stream);    // += basis@sw1^T

    spline_basis_kernel<<<cdiv((long)cN * cFFN, 256), 256, 0, stream>>>(y1, basisF, (long)cN * cFFN);
    run_gemm<1, 0, 0>(y1, f2b, nullptr, y2, cN, cHID, cFFN, stream);              // silu(y1)@base2^T
    run_gemm<0, 2, 0>(basisF, sw2, nullptr, y2, cN, cHID, cFFN * cSK, stream);    // += basis@sw2^T

    residual_ln_kernel<<<cdiv(cN, 8), 256, 0, stream>>>(h, y2, norm2_w + l * cHID, norm2_b + l * cHID);
  }

  // ---- readout: per-graph mean/max pooling + LN ---------------------------
  hipMemsetAsync(cnt, 0, (size_t)cNG * 4, stream);
  hipMemsetAsync(sums, 0, (size_t)cNG * cHID * 4, stream);
  fill_f32_kernel<<<cdiv((long)cNG * cHID, 256), 256, 0, stream>>>(mx, -1e30f, (long)cNG * cHID);
  readout_acc_kernel<<<cdiv(cN, 8), 256, 0, stream>>>(h, batch, sums, mx, cnt);
  readout_norm_kernel<<<cNG, 256, 0, stream>>>(sums, mx, cnt, ro_norm_w, ro_norm_b, gcat);

  // ---- head KAN 1: [64,256] -> [64,128] -----------------------------------
  make_sw_kernel<<<cdiv((long)cKH * 256 * cSK, 256), 256, 0, stream>>>(h1_spw, h1_sc, swh1, cKH, 256);
  spline_basis_kernel<<<cdiv((long)cNG * 256, 256), 256, 0, stream>>>(gcat, bgin, (long)cNG * 256);
  run_gemm<1, 0, 0>(gcat, h1_base, nullptr, g2, cNG, cKH, 256, stream);
  run_gemm<0, 2, 0>(bgin, swh1, nullptr, g2, cNG, cKH, 256 * cSK, stream);

  // ---- head KAN 2: [64,128] -> [64,10] ------------------------------------
  make_sw_kernel<<<cdiv((long)cNC * cKH * cSK, 256), 256, 0, stream>>>(h2_spw, h2_sc, swh2, cNC, cKH);
  spline_basis_kernel<<<cdiv((long)cNG * cKH, 256), 256, 0, stream>>>(g2, bg2, (long)cNG * cKH);
  run_gemm<1, 0, 0>(g2, h2_base, nullptr, out, cNG, cNC, cKH, stream);
  run_gemm<0, 2, 0>(bg2, swh2, nullptr, out, cNG, cNC, cKH * cSK, stream);
}
